// SelfAttentionHead_9156870275342
// MI455X (gfx1250) — matrix operand order repaired and checked by reading, not compiled, not run on hardware
//
#include <hip/hip_runtime.h>

// B=4, T=4096, D=2048, H=128
#define TT   4096
#define DD   2048
#define HH   128

typedef __attribute__((ext_vector_type(16))) __bf16 v16bf;
typedef __attribute__((ext_vector_type(4)))  __bf16 v4bf;
typedef __attribute__((ext_vector_type(8)))  float  v8f;
typedef __attribute__((ext_vector_type(8)))  unsigned int v8u;

// ws layout (bytes)
#define WT_OFF  0                       // 3*128*2048*2 = 1572864
#define Q_OFF   1572864                 // 16384*128*2  = 4194304
#define K_OFF   (Q_OFF + 4194304)
#define VT_OFF  (K_OFF + 4194304)

// Load a 16-element bf16 fragment from two 16-byte chunks.
static __device__ __forceinline__ v16bf ld16(const __bf16* p0, const __bf16* p1) {
    v8u t;
    const uint4 a = *reinterpret_cast<const uint4*>(p0);
    const uint4 c = *reinterpret_cast<const uint4*>(p1);
    t[0] = a.x; t[1] = a.y; t[2] = a.z; t[3] = a.w;
    t[4] = c.x; t[5] = c.y; t[6] = c.z; t[7] = c.w;
    return __builtin_bit_cast(v16bf, t);
}

static __device__ __forceinline__ float half_max(float v) {
    v = fmaxf(v, __shfl_xor(v, 1, 32));
    v = fmaxf(v, __shfl_xor(v, 2, 32));
    v = fmaxf(v, __shfl_xor(v, 4, 32));
    v = fmaxf(v, __shfl_xor(v, 8, 32));
    return v;
}
static __device__ __forceinline__ float half_sum(float v) {
    v += __shfl_xor(v, 1, 32);
    v += __shfl_xor(v, 2, 32);
    v += __shfl_xor(v, 4, 32);
    v += __shfl_xor(v, 8, 32);
    return v;
}

// ---------------------------------------------------------------------------
// Kernel 1: convert Wq/Wk/Wv (f32, [D,H] row-major) -> bf16 col-major Wt[n][k]
// ---------------------------------------------------------------------------
__global__ __launch_bounds__(256) void convert_weights(
    const float* __restrict__ Wq, const float* __restrict__ Wk,
    const float* __restrict__ Wv, __bf16* __restrict__ wt) {
    int idx = blockIdx.x * 256 + threadIdx.x;      // < 3*128*2048 = 786432
    if (idx >= 3 * HH * DD) return;
    int mat = idx >> 18;                           // / (128*2048)
    int rem = idx & (HH * DD - 1);
    int n   = rem >> 11;                           // / 2048
    int kk  = rem & (DD - 1);
    const float* W = (mat == 0) ? Wq : (mat == 1) ? Wk : Wv;
    wt[idx] = (__bf16)W[(size_t)kk * HH + n];
}

// ---------------------------------------------------------------------------
// Kernel 2: fused QKV projection.  Block = 384 threads = 12 waves.
// wave w: mat = w/4 (0=q,1=k,2=v), stripe = w%4 (16 rows each of a 64-row tile)
// ---------------------------------------------------------------------------
__global__ __launch_bounds__(384) void qkv_proj(
    const float* __restrict__ x, const __bf16* __restrict__ wt,
    const float* __restrict__ bq, const float* __restrict__ bk,
    const float* __restrict__ bv,
    __bf16* __restrict__ qo, __bf16* __restrict__ ko, __bf16* __restrict__ vto) {

    __shared__ __align__(16) __bf16 xs[64 * 32];   // 4 KB bf16 x-tile

    const int tid    = threadIdx.x;
    const int lane   = tid & 31;
    const int w      = tid >> 5;
    const int mat    = w >> 2;                     // 0..2
    const int stripe = w & 3;                      // 0..3
    const int rowbase = blockIdx.x * 64;
    const int ln15 = lane & 15;
    const int hi   = lane >> 4;

    const __bf16* wmat = wt + (size_t)mat * HH * DD;

    v8f acc[8];
#pragma unroll
    for (int t = 0; t < 8; ++t) acc[t] = v8f{};

    // staging indices: 512 float4 quads per 64x32 tile; quad -> (row, col4)
    const int q0row = tid >> 3,          q0c = (tid & 7) * 4;            // quads 0..383
    const int q1row = (tid + 384) >> 3,  q1c = ((tid + 384) & 7) * 4;    // quads 384..511

    for (int kb = 0; kb < DD; kb += 32) {
        // vectorized f32->bf16 staging: float4 loads, b64 LDS stores
        {
            float4 f0 = *reinterpret_cast<const float4*>(
                &x[(size_t)(rowbase + q0row) * DD + kb + q0c]);
            v4bf h0 = { (__bf16)f0.x, (__bf16)f0.y, (__bf16)f0.z, (__bf16)f0.w };
            *reinterpret_cast<v4bf*>(&xs[q0row * 32 + q0c]) = h0;
            if (tid < 128) {
                float4 f1 = *reinterpret_cast<const float4*>(
                    &x[(size_t)(rowbase + q1row) * DD + kb + q1c]);
                v4bf h1 = { (__bf16)f1.x, (__bf16)f1.y, (__bf16)f1.z, (__bf16)f1.w };
                *reinterpret_cast<v4bf*>(&xs[q1row * 32 + q1c]) = h1;
            }
        }
        if (kb + 32 < DD)   // warm caches for next tile (global_prefetch_b8)
            __builtin_prefetch(&x[(size_t)(rowbase + (tid & 63)) * DD + kb + 32], 0, 1);
        __syncthreads();

        // A fragment: 16-bit A 16x32 layout (two 16B runs at hi*8 and 16+hi*8)
        const int arow = stripe * 16 + ln15;
        const __bf16* ap = &xs[arow * 32 + hi * 8];
        v16bf afrag = ld16(ap, ap + 16);

        // preload all 8 B fragments, then issue all 8 WMMAs (overlapped loads)
        v16bf bfrag[8];
#pragma unroll
        for (int t = 0; t < 8; ++t) {
            const __bf16* bp = wmat + (size_t)(t * 16 + ln15) * DD + kb + hi * 8;
            bfrag[t] = ld16(bp, bp + 16);
        }
#pragma unroll
        for (int t = 0; t < 8; ++t) {
            acc[t] = __builtin_amdgcn_wmma_f32_16x16x32_bf16(
                false, afrag, false, bfrag[t], (short)0, acc[t], false, false);
        }
        __syncthreads();
    }

    // epilogue: bias add + store bf16 (v stored transposed)
    const float* bias = (mat == 0) ? bq : (mat == 1) ? bk : bv;
#pragma unroll
    for (int t = 0; t < 8; ++t) {
        int col = t * 16 + ln15;
        float bsv = bias[col];
#pragma unroll
        for (int r = 0; r < 8; ++r) {
            int row = rowbase + stripe * 16 + r + hi * 8;   // global row in [0,16384)
            float val = acc[t][r] + bsv;
            if (mat == 0) {
                qo[(size_t)row * HH + col] = (__bf16)val;
            } else if (mat == 1) {
                ko[(size_t)row * HH + col] = (__bf16)val;
            } else {
                int b  = row >> 12;            // /4096
                int tt = row & (TT - 1);
                vto[((size_t)b * HH + col) * TT + tt] = (__bf16)val;
            }
        }
    }
}

// ---------------------------------------------------------------------------
// Kernel 3: causal flash attention.  Block = 128 threads = 4 waves.
// One wave owns one 16-query tile; online softmax in registers.
// ---------------------------------------------------------------------------
__global__ __launch_bounds__(128) void flash_attn(
    const __bf16* __restrict__ q, const __bf16* __restrict__ k,
    const __bf16* __restrict__ vt, float* __restrict__ out) {

    __shared__ __align__(16) __bf16 ps[4][16 * 32];    // per-wave P bounce (1 KB each)

    const int tid  = threadIdx.x;
    const int lane = tid & 31;
    const int w    = tid >> 5;
    const int tile  = blockIdx.x * 4 + w;              // 0..1023
    const int b     = tile >> 8;                       // 256 tiles per batch
    const int qbase = (tile & 255) << 4;
    const int ln15 = lane & 15;
    const int hi   = lane >> 4;
    const int hi8  = hi * 8;
    const size_t bT = (size_t)b * TT;

    // Q fragments over H=128: 4 chunks of K=32 (A-frag: runs at hi*8, 16+hi*8)
    v16bf qf[4];
#pragma unroll
    for (int c = 0; c < 4; ++c) {
        const __bf16* p = q + (bT + qbase + ln15) * HH + c * 32 + hi * 8;
        qf[c] = ld16(p, p + 16);
    }

    v8f o[8];
#pragma unroll
    for (int t = 0; t < 8; ++t) o[t] = v8f{};
    float M[8], L[8];
#pragma unroll
    for (int r = 0; r < 8; ++r) { M[r] = -1e30f; L[r] = 0.0f; }

    const float scale = 0.08838834764831845f;          // 1/sqrt(128)
    const int qend = qbase + 15;
    __bf16* pl = &ps[w][0];

    for (int kb = 0; kb <= qend; kb += 32) {
        // ---- S = Q K^T for two 16-key subtiles: preload all 8 K frags first
        v16bf kf[8];
#pragma unroll
        for (int sub = 0; sub < 2; ++sub) {
            int krow = kb + sub * 16 + ln15;
            if (krow > TT - 1) krow = TT - 1;          // masked anyway
#pragma unroll
            for (int c = 0; c < 4; ++c) {
                const __bf16* p = k + (bT + krow) * HH + c * 32 + hi * 8;
                kf[sub * 4 + c] = ld16(p, p + 16);
            }
        }
        v8f s[2];
#pragma unroll
        for (int sub = 0; sub < 2; ++sub) {
            s[sub] = v8f{};
#pragma unroll
            for (int c = 0; c < 4; ++c) {
                s[sub] = __builtin_amdgcn_wmma_f32_16x16x32_bf16(
                    false, qf[c], false, kf[sub * 4 + c], (short)0, s[sub], false, false);
            }
        }

        // scale + causal mask (C-layout: lane half hi holds rows r+hi*8)
#pragma unroll
        for (int sub = 0; sub < 2; ++sub) {
#pragma unroll
            for (int r = 0; r < 8; ++r) {
                int qi = qbase + r + hi8;
                int kj = kb + sub * 16 + ln15;
                float sv = s[sub][r] * scale;
                s[sub][r] = (kj <= qi) ? sv : -1e30f;
            }
        }

        // online softmax
        float alpha[8], rs[8];
#pragma unroll
        for (int r = 0; r < 8; ++r) {
            float rm = half_max(fmaxf(s[0][r], s[1][r]));
            float mn = fmaxf(M[r], rm);
            alpha[r] = __expf(M[r] - mn);
            M[r] = mn;
            rs[r] = 0.0f;
        }
#pragma unroll
        for (int sub = 0; sub < 2; ++sub) {
#pragma unroll
            for (int r = 0; r < 8; ++r) {
                float p = __expf(s[sub][r] - M[r]);
                rs[r] += p;
                pl[(r + hi8) * 32 + sub * 16 + ln15] = (__bf16)p;  // C->A bounce
            }
        }
#pragma unroll
        for (int r = 0; r < 8; ++r) {
            L[r] = L[r] * alpha[r] + half_sum(rs[r]);
        }
#pragma unroll
        for (int t = 0; t < 8; ++t)
#pragma unroll
            for (int r = 0; r < 8; ++r) o[t][r] *= alpha[r];

        // wave-local LDS RAW: stores/loads are in-order per wave; fence compiler
        asm volatile("s_wait_dscnt 0" ::: "memory");

        // P as A fragment (16 queries x 32 keys)
        const __bf16* pp = &pl[ln15 * 32 + hi * 8];
        v16bf pf = ld16(pp, pp + 16);

        // ---- O += P * V: preload all 8 V frags, then 8 WMMAs
        int kv = kb + hi * 8; if (kv > TT - 24) kv = TT - 24;      // masked anyway
        v16bf vf[8];
#pragma unroll
        for (int t = 0; t < 8; ++t) {
            const __bf16* p = vt + ((size_t)b * HH + t * 16 + ln15) * TT + kv;
            vf[t] = ld16(p, p + 16);
        }
#pragma unroll
        for (int t = 0; t < 8; ++t) {
            o[t] = __builtin_amdgcn_wmma_f32_16x16x32_bf16(
                false, pf, false, vf[t], (short)0, o[t], false, false);
        }
    }

    // epilogue: normalize, store f32 [B,T,H]
#pragma unroll
    for (int t = 0; t < 8; ++t) {
#pragma unroll
        for (int r = 0; r < 8; ++r) {
            out[(bT + qbase + r + hi8) * HH + t * 16 + ln15] = o[t][r] / L[r];
        }
    }
}

// ---------------------------------------------------------------------------
extern "C" void kernel_launch(void* const* d_in, const int* in_sizes, int n_in,
                              void* d_out, int out_size, void* d_ws, size_t ws_size,
                              hipStream_t stream) {
    (void)in_sizes; (void)n_in; (void)out_size; (void)ws_size;
    const float* x  = (const float*)d_in[0];
    const float* Wk = (const float*)d_in[1];
    const float* bk = (const float*)d_in[2];
    const float* Wq = (const float*)d_in[3];
    const float* bq = (const float*)d_in[4];
    const float* Wv = (const float*)d_in[5];
    const float* bv = (const float*)d_in[6];

    char* ws = (char*)d_ws;
    __bf16* wt   = (__bf16*)(ws + WT_OFF);
    __bf16* qbuf = (__bf16*)(ws + Q_OFF);
    __bf16* kbuf = (__bf16*)(ws + K_OFF);
    __bf16* vtb  = (__bf16*)(ws + VT_OFF);

    convert_weights<<<3072, 256, 0, stream>>>(Wq, Wk, Wv, wt);
    qkv_proj<<<256, 384, 0, stream>>>(x, wt, bq, bk, bv, qbuf, kbuf, vtb);
    flash_attn<<<256, 128, 0, stream>>>(qbuf, kbuf, vtb, (float*)d_out);
}
